// Attention_1623497638729
// MI455X (gfx1250) — matrix operand order repaired and checked by reading, not compiled, not run on hardware
//
#include <hip/hip_runtime.h>

// ---------------------------------------------------------------------------
// MHA for MI455X (gfx1250): bf16 WMMA GEMMs + block-cooperative flash
// attention. Matrix math: v_wmma_f32_16x16x32_bf16 (f32 accum).
// Data movement: GEMMs use double-buffered global_load_async_to_lds_b128
// (ASYNCcnt); attention K/V tiles use the Tensor Data Mover
// (tensor_load_to_lds, TENSORcnt) with LDS row padding done by the TDM.
// ---------------------------------------------------------------------------

typedef __attribute__((ext_vector_type(16))) __bf16 v16bf;
typedef __attribute__((ext_vector_type(4)))  __bf16 v4bf;
typedef __attribute__((ext_vector_type(8)))  float  v8f;
typedef unsigned int u32x4 __attribute__((ext_vector_type(4)));
typedef int          i32x8 __attribute__((ext_vector_type(8)));
typedef int          i32x4 __attribute__((ext_vector_type(4)));

#define DIMX   768
#define HEADS  12
#define DHEAD  64
#define QKV3   2304          // 3*768
#define BATCH  4
#define SEQ    2048
#define ROWS   (BATCH*SEQ)   // 8192

#if defined(__has_builtin)
#if __has_builtin(__builtin_amdgcn_tensor_load_to_lds)
#define HAVE_TDM 1
#endif
#endif
#ifndef HAVE_TDM
#define HAVE_TDM 0
#endif

__device__ __forceinline__ v8f wmma_bf16(v16bf a, v16bf b, v8f c) {
    return __builtin_amdgcn_wmma_f32_16x16x32_bf16(false, a, false, b,
                                                   (short)0, c, false, false);
}

// async 16B copy global -> LDS (VFLAT GV mode, tracked by ASYNCcnt)
__device__ __forceinline__ void async_b128(unsigned lds_addr,
                                           unsigned long long gaddr) {
    asm volatile("global_load_async_to_lds_b128 %0, %1, off"
                 :: "v"(lds_addr), "v"(gaddr) : "memory");
}
__device__ __forceinline__ unsigned lds_of(const void* p) {
    return (unsigned)(size_t)p;      // flat LDS addr: hw uses addr[31:0]
}
#define WAIT_ASYNC_LIT(n) asm volatile("s_wait_asynccnt " #n ::: "memory")

#if HAVE_TDM
// TDM 2D tile load: rows x cols bf16 tile from a row-strided tensor into LDS.
// Pads each 64-elem (32-DWORD) row with 4 DWORDs -> 72-elem LDS rows (KV_LDP).
__device__ __forceinline__ void tdm_load_2d_kv(unsigned lds_addr,
                                               const void* gptr,
                                               int rows, int cols,
                                               int row_stride_elems)
{
    unsigned long long ga = (unsigned long long)gptr;
    u32x4 g0;
    g0[0] = 1u;                                     // count=1 (valid user D#)
    g0[1] = lds_addr;                               // LDS byte address
    g0[2] = (unsigned)ga;                           // global_addr[31:0]
    g0[3] = (unsigned)((ga >> 32) & 0x01FFFFFFull)  // global_addr[56:32]
          | (2u << 30);                             // type = 2 ("image")
    i32x8 g1;
    g1[0] = (1 << 16)                               // data_size = 2 bytes
          | (1 << 20)                               // pad_enable
          | (4 << 22)                               // pad_interval: 32 DWORDs
          | (3 << 25);                              // pad_amount: 4 DWORDs
    g1[1] = (cols & 0xFFFF) << 16;                  // tensor_dim0[15:0]
    g1[2] = (cols >> 16) | ((rows & 0xFFFF) << 16); // dim0[31:16] | dim1[15:0]
    g1[3] = (rows >> 16) | (cols << 16);            // dim1[31:16] | tile_dim0
    g1[4] = rows;                                   // tile_dim1 (tile_dim2=0)
    g1[5] = row_stride_elems;                       // tensor_dim0_stride[31:0]
    g1[6] = 0;                                      // stride hi | dim1_stride lo
    g1[7] = 0;
    i32x4 gz = {0, 0, 0, 0};
#if __has_include(<hip/amd_detail/amd_gfx1250_TDM.h>)
    i32x8 gz8 = {0, 0, 0, 0, 0, 0, 0, 0};           // clang-23 / therock: 6 args
    __builtin_amdgcn_tensor_load_to_lds(g0, g1, gz, gz, gz8, 0);
#else
    __builtin_amdgcn_tensor_load_to_lds(g0, g1, gz, gz, 0);  // ROCm 7.2: 5 args
#endif
}
#endif  // HAVE_TDM

// A-fragment (16x32 bf16, row-major, ld): lane halves K 0-15/16-31.
__device__ __forceinline__ v16bf load_a_frag(const __bf16* base, int ld) {
    const int lane = threadIdx.x & 31;
    const int m  = lane & 15;
    const int kb = (lane >> 4) << 4;            // 0 or 16
    v16bf a;
#pragma unroll
    for (int e = 0; e < 16; ++e) {
        a[e] = base[(size_t)m * ld + kb + e];
    }
    return a;
}
// B-fragment (32x16 bf16 from row-major [K x N], ld): lane halves K 0-15/16-31.
__device__ __forceinline__ v16bf load_b_frag(const __bf16* base, int ld) {
    const int lane = threadIdx.x & 31;
    const int n  = lane & 15;
    const int kb = (lane >> 4) << 4;            // 0 or 16
    v16bf b;
#pragma unroll
    for (int e = 0; e < 16; ++e)
        b[e] = base[(size_t)(kb + e) * ld + n];
    return b;
}

// ---------------------------------------------------------------------------
// f32 -> bf16 (4 elems/thread)
// ---------------------------------------------------------------------------
__global__ void cvt_f32_bf16(const float* __restrict__ src,
                             __bf16* __restrict__ dst, int n4) {
    int i = blockIdx.x * blockDim.x + threadIdx.x;
    int stride = gridDim.x * blockDim.x;
    for (; i < n4; i += stride) {
        float4 f = ((const float4*)src)[i];
        v4bf o;
        o[0] = (__bf16)f.x; o[1] = (__bf16)f.y;
        o[2] = (__bf16)f.z; o[3] = (__bf16)f.w;
        ((v4bf*)dst)[i] = o;
    }
}

// ---------------------------------------------------------------------------
// bf16 GEMM, 128x128 block tile, 128 threads = 4 waves (each 64x64 = 4x4 WMMA
// tiles), K-step 32, double-buffered async global->LDS staging.
// Requires M%128==0, N%128==0, K%32==0 (true for all calls here).
// ---------------------------------------------------------------------------
#define LDA_P 48    // 32+16 pad; row = 96B  (16B multiple for async B128)
#define LDB_P 136   // 128+8 pad; row = 272B (16B multiple)

__device__ __forceinline__ void gemm_stage(
    const __bf16* __restrict__ A, const __bf16* __restrict__ B,
    __bf16* as, __bf16* bs, int row0, int col0, int k0, int K, int N, int t)
{
#pragma unroll
    for (int i = 0; i < 4; ++i) {              // A: 128x32 = 512 x 8-elem chunks
        int c = i * 128 + t;
        int r = c >> 2, col = (c & 3) * 8;
        async_b128(lds_of(&as[r * LDA_P + col]),
                   (unsigned long long)&A[(size_t)(row0 + r) * K + k0 + col]);
    }
#pragma unroll
    for (int i = 0; i < 4; ++i) {              // B: 32x128
        int c = i * 128 + t;
        int r = c >> 4, col = (c & 15) * 8;
        async_b128(lds_of(&bs[r * LDB_P + col]),
                   (unsigned long long)&B[(size_t)(k0 + r) * N + col0 + col]);
    }
}

template <int OUT_F32>
__global__ __launch_bounds__(128) void gemm_bf16(
    const __bf16* __restrict__ A, const __bf16* __restrict__ B,
    void* __restrict__ Cout, const float* __restrict__ bias,
    int M, int N, int K)
{
    __shared__ __bf16 As[2][128 * LDA_P];
    __shared__ __bf16 Bs[2][32 * LDB_P];

    const int t    = threadIdx.x;
    const int lane = t & 31;
    const int wid  = t >> 5;
    const int wy   = wid >> 1, wx = wid & 1;
    const int row0 = blockIdx.y * 128;
    const int col0 = blockIdx.x * 128;

    const v8f vzero = {0.f,0.f,0.f,0.f,0.f,0.f,0.f,0.f};
    v8f acc[4][4];
#pragma unroll
    for (int i = 0; i < 4; ++i)
#pragma unroll
        for (int j = 0; j < 4; ++j) acc[i][j] = vzero;

    const int nk = K / 32;
    gemm_stage(A, B, As[0], Bs[0], row0, col0, 0, K, N, t);

    for (int kt = 0; kt < nk; ++kt) {
        const int cur = kt & 1;
        if (kt + 1 < nk) {
            gemm_stage(A, B, As[1 - cur], Bs[1 - cur],
                       row0, col0, (kt + 1) * 32, K, N, t);
            WAIT_ASYNC_LIT(8);   // 8 in-flight prefetch ops; current tile done
        } else {
            WAIT_ASYNC_LIT(0);
        }
        __syncthreads();

        v16bf af[4], bfr[4];
#pragma unroll
        for (int ii = 0; ii < 4; ++ii)
            af[ii] = load_a_frag(&As[cur][(wy * 64 + ii * 16) * LDA_P], LDA_P);
#pragma unroll
        for (int jj = 0; jj < 4; ++jj)
            bfr[jj] = load_b_frag(&Bs[cur][wx * 64 + jj * 16], LDB_P);
#pragma unroll
        for (int ii = 0; ii < 4; ++ii)
#pragma unroll
            for (int jj = 0; jj < 4; ++jj)
                acc[ii][jj] = wmma_bf16(af[ii], bfr[jj], acc[ii][jj]);

        __syncthreads();   // all waves done with `cur` before it is re-staged
    }

    // epilogue: C layout -> col = lane&15, row = vgpr + 8*(lane>>4)
    const int cn   = lane & 15;
    const int rhib = (lane >> 4) << 3;
#pragma unroll
    for (int jj = 0; jj < 4; ++jj) {
        const int col = col0 + wx * 64 + jj * 16 + cn;
        const float bb = OUT_F32 ? bias[col] : 0.f;
#pragma unroll
        for (int ii = 0; ii < 4; ++ii) {
#pragma unroll
            for (int r = 0; r < 8; ++r) {
                const int row = row0 + wy * 64 + ii * 16 + rhib + r;
                const float v = acc[ii][jj][r];
                if (OUT_F32)
                    ((float*)Cout)[(size_t)row * N + col] = v + bb;
                else
                    ((__bf16*)Cout)[(size_t)row * N + col] = (__bf16)v;
            }
        }
    }
}

// ---------------------------------------------------------------------------
// Flash attention. Block = 128 threads (4 waves) on one (b,h); each wave owns
// a 16-query tile (64 q rows / block). K/V 32x64 tiles staged ONCE per block
// into double-buffered LDS: TDM tensor_load_to_lds (wave 0, TENSORcnt) with
// hardware row padding; async-B128 fallback if the builtin is unavailable.
// Online softmax with shfl_xor row reductions; P routed C-layout -> LDS ->
// A-layout; PV accumulated in 4 f32 fragments (16x64). ctx in b n (h d).
// ---------------------------------------------------------------------------
#define KV_LDP 72   // 64+8 pad; row = 144B (matches TDM pad_interval/amount)

__global__ __launch_bounds__(128) void attn_flash(
    const __bf16* __restrict__ qkv, __bf16* __restrict__ ctx)
{
    __shared__ __bf16 Ks[2][32 * KV_LDP];
    __shared__ __bf16 Vs[2][32 * KV_LDP];
    __shared__ __bf16 pbuf[4][16 * 40];

    const int t = threadIdx.x, lane = t & 31, wid = t >> 5;
    const int bh = blockIdx.x, b = bh / HEADS, h = bh % HEADS;
    const int q0 = (blockIdx.y * 4 + wid) * 16;
    const size_t rowbase = (size_t)b * SEQ;

    const __bf16* Qp    = qkv + (rowbase + q0) * QKV3 + h * DHEAD;
    const __bf16* Kbase = qkv + rowbase * QKV3 + DIMX     + h * DHEAD;
    const __bf16* Vbase = qkv + rowbase * QKV3 + 2 * DIMX + h * DHEAD;

    auto stage = [&](int j, int buf) {
#if HAVE_TDM
        if (wid == 0) {   // one TDM descriptor pair per tile, whole block shares
            tdm_load_2d_kv(lds_of(&Ks[buf][0]), &Kbase[(size_t)j * QKV3],
                           32, 64, QKV3);
            tdm_load_2d_kv(lds_of(&Vs[buf][0]), &Vbase[(size_t)j * QKV3],
                           32, 64, QKV3);
        }
#else
#pragma unroll
        for (int i = 0; i < 2; ++i) {
            int c = i * 128 + t;                 // 0..255
            int r = c >> 3, col = (c & 7) * 8;
            async_b128(lds_of(&Ks[buf][r * KV_LDP + col]),
                       (unsigned long long)&Kbase[(size_t)(j + r) * QKV3 + col]);
            async_b128(lds_of(&Vs[buf][r * KV_LDP + col]),
                       (unsigned long long)&Vbase[(size_t)(j + r) * QKV3 + col]);
        }
#endif
    };

    const v16bf qa0 = load_a_frag(Qp,      QKV3);   // d = 0..31
    const v16bf qa1 = load_a_frag(Qp + 32, QKV3);   // d = 32..63

    const v8f vzero = {0.f,0.f,0.f,0.f,0.f,0.f,0.f,0.f};
    v8f acc[4];
    float mrun[8], lrun[8];
#pragma unroll
    for (int d = 0; d < 4; ++d) acc[d] = vzero;
#pragma unroll
    for (int r = 0; r < 8; ++r) { mrun[r] = -__builtin_inff(); lrun[r] = 0.f; }

    const int cn   = lane & 15;
    const int hi   = lane >> 4;
    const int kb16 = hi << 4;
    const float scale = 0.125f;          // 1/sqrt(64)
    __bf16* pb = &pbuf[wid][0];

    stage(0, 0);
    for (int jt = 0; jt < SEQ / 32; ++jt) {
        const int cur = jt & 1;
        if (jt + 1 < SEQ / 32) {
            stage((jt + 1) * 32, 1 - cur);
#if HAVE_TDM
            if (wid == 0) __builtin_amdgcn_s_wait_tensorcnt((short)2);
#else
            WAIT_ASYNC_LIT(4);
#endif
        } else {
#if HAVE_TDM
            if (wid == 0) __builtin_amdgcn_s_wait_tensorcnt((short)0);
#else
            WAIT_ASYNC_LIT(0);
#endif
        }
        __syncthreads();
        const __bf16* ks = &Ks[cur][0];
        const __bf16* vs = &Vs[cur][0];

        // K^T B-fragments from LDS (lane = key col, elem = d, contiguous)
        v16bf kb0lo, kb0hi, kb1lo, kb1hi;
#pragma unroll
        for (int e = 0; e < 16; ++e) {
            kb0lo[e] = ks[cn * KV_LDP + kb16 + e];
            kb0hi[e] = ks[cn * KV_LDP + 32 + kb16 + e];
            kb1lo[e] = ks[(16 + cn) * KV_LDP + kb16 + e];
            kb1hi[e] = ks[(16 + cn) * KV_LDP + 32 + kb16 + e];
        }
        v8f s0 = wmma_bf16(qa0, kb0lo, vzero);
        s0     = wmma_bf16(qa1, kb0hi, s0);
        v8f s1 = wmma_bf16(qa0, kb1lo, vzero);
        s1     = wmma_bf16(qa1, kb1hi, s1);

        // online softmax over 32 new keys
#pragma unroll
        for (int r = 0; r < 8; ++r) {
            float e0 = s0[r] * scale, e1 = s1[r] * scale;
            float mx = fmaxf(e0, e1);
            mx = fmaxf(mx, __shfl_xor(mx, 1));
            mx = fmaxf(mx, __shfl_xor(mx, 2));
            mx = fmaxf(mx, __shfl_xor(mx, 4));
            mx = fmaxf(mx, __shfl_xor(mx, 8));
            float mnew = fmaxf(mrun[r], mx);
            float corr = __expf(mrun[r] - mnew);
            float p0 = __expf(e0 - mnew);
            float p1 = __expf(e1 - mnew);
            float rs = p0 + p1;
            rs += __shfl_xor(rs, 1);
            rs += __shfl_xor(rs, 2);
            rs += __shfl_xor(rs, 4);
            rs += __shfl_xor(rs, 8);
            lrun[r] = lrun[r] * corr + rs;
            mrun[r] = mnew;
            acc[0][r] *= corr; acc[1][r] *= corr;
            acc[2][r] *= corr; acc[3][r] *= corr;
            pb[(r + (hi << 3)) * 40 + cn]      = (__bf16)p0;
            pb[(r + (hi << 3)) * 40 + 16 + cn] = (__bf16)p1;
        }

#if __has_builtin(__builtin_amdgcn_s_wait_dscnt)
        __builtin_amdgcn_s_wait_dscnt(0);
#else
        asm volatile("s_wait_dscnt 0" ::: "memory");
#endif
        v16bf pa = load_a_frag(pb, 40);   // 16x32 A-fragment of P

#pragma unroll
        for (int db = 0; db < 4; ++db) {
            v16bf vb;
#pragma unroll
            for (int e = 0; e < 16; ++e)
                vb[e] = vs[(kb16 + e) * KV_LDP + db * 16 + cn];
            acc[db] = wmma_bf16(pa, vb, acc[db]);
        }
        __syncthreads();   // all waves done with `cur` tiles before re-stage
    }

    float linv[8];
#pragma unroll
    for (int r = 0; r < 8; ++r) linv[r] = 1.f / lrun[r];
    __bf16* cp = ctx + (rowbase + q0) * DIMX + h * DHEAD;
#pragma unroll
    for (int db = 0; db < 4; ++db)
#pragma unroll
        for (int r = 0; r < 8; ++r)
            cp[(size_t)(r + (hi << 3)) * DIMX + db * 16 + cn] =
                (__bf16)(acc[db][r] * linv[r]);
}

// ---------------------------------------------------------------------------
extern "C" void kernel_launch(void* const* d_in, const int* in_sizes, int n_in,
                              void* d_out, int out_size, void* d_ws, size_t ws_size,
                              hipStream_t stream)
{
    const float* x    = (const float*)d_in[0];
    const float* Wqkv = (const float*)d_in[1];
    const float* Wout = (const float*)d_in[2];
    const float* bout = (const float*)d_in[3];

    char* ws = (char*)d_ws;
    size_t off = 0;
    __bf16* xb    = (__bf16*)(ws + off); off += (size_t)ROWS * DIMX * 2;
    __bf16* wqkvb = (__bf16*)(ws + off); off += (size_t)DIMX * QKV3 * 2;
    __bf16* woutb = (__bf16*)(ws + off); off += (size_t)DIMX * DIMX * 2;
    __bf16* qkvb  = (__bf16*)(ws + off); off += (size_t)ROWS * QKV3 * 2;
    __bf16* ctxb  = xb;   // reuse: xb dead after QKV GEMM

    cvt_f32_bf16<<<dim3(1024), 256, 0, stream>>>(x,    xb,    ROWS * DIMX / 4);
    cvt_f32_bf16<<<dim3(256),  256, 0, stream>>>(Wqkv, wqkvb, DIMX * QKV3 / 4);
    cvt_f32_bf16<<<dim3(256),  256, 0, stream>>>(Wout, woutb, DIMX * DIMX / 4);

    // QKV = x * Wqkv  -> [ROWS, 2304] bf16
    gemm_bf16<0><<<dim3(QKV3 / 128, ROWS / 128), 128, 0, stream>>>(
        xb, wqkvb, qkvb, nullptr, ROWS, QKV3, DIMX);

    // flash attention -> ctx [ROWS, 768] bf16 in b n (h d)
    attn_flash<<<dim3(BATCH * HEADS, SEQ / 64), 128, 0, stream>>>(qkvb, ctxb);

    // out = ctx * Wout + bout -> f32
    gemm_bf16<1><<<dim3(DIMX / 128, ROWS / 128), 128, 0, stream>>>(
        ctxb, woutb, d_out, bout, ROWS, DIMX, DIMX);
}
